// DNRI_Decoder_56487409877020
// MI455X (gfx1250) — matrix operand order repaired and checked by reading, not compiled, not run on hardware
//
#include <hip/hip_runtime.h>
#include <hip/hip_bf16.h>
#include <math.h>
#include <stdint.h>

#define DEV __device__ __forceinline__

typedef __attribute__((ext_vector_type(16))) _Float16 v16h;
typedef __attribute__((ext_vector_type(8)))  _Float16 v8h;
typedef __attribute__((ext_vector_type(8)))  float    v8f;

constexpr int B  = 64, N = 100, H = 64, IN = 32;
constexpr int E  = N * (N - 1);     // 9900
constexpr int NPAD = 112;           // senders padded to 7*16
constexpr int LSTR = 72;            // LDS row stride (halves), conflict-free for b128 reads

// ---- f16 weight scratch layout (half-element offsets) ----
constexpr int HO_W1   = 0;                     // W1catT [128c][64k]  (c<64 -> RECV part, c>=64 -> SEND part)
constexpr int HO_W2   = HO_W1  + 128 * 64;     // W2T    [64n][64k]
constexpr int HO_WHR  = HO_W2  + 64 * 64;
constexpr int HO_WHI  = HO_WHR + 64 * 64;
constexpr int HO_WHH  = HO_WHI + 64 * 64;
constexpr int HO_WIR  = HO_WHH + 64 * 64;      // [64n][32k]
constexpr int HO_WII  = HO_WIR + 64 * 32;
constexpr int HO_WIN  = HO_WII + 64 * 32;
constexpr int HO_WM1  = HO_WIN + 64 * 32;
constexpr int HO_WM2  = HO_WM1 + 64 * 64;
constexpr int HO_PROJ = HO_WM2 + 64 * 64;      // 3 x [112n][6400k], n>=100 zero-padded
constexpr int H_TOTAL = HO_PROJ + 3 * NPAD * (N * H);

// ---- float scratch layout (byte offsets) ----
constexpr size_t OFF_GATE = 0;                                          // [B*E] f32 (= edges[:,:,1])
constexpr size_t OFF_X    = OFF_GATE + sizeof(float) * (size_t)B * E;   // [B][112][64] f32 (h_i @ W1_bot), pad rows = 0
constexpr size_t OFF_Y    = OFF_X  + sizeof(float) * (size_t)B * NPAD * H; // [B][100][64] f32 (h_j @ W1_top + b1)
constexpr size_t OFF_AGG  = OFF_Y  + sizeof(float) * (size_t)B * N * H;
constexpr size_t OFF_FLAT = OFF_AGG + sizeof(float) * (size_t)B * N * H;   // [B][6400] f16 pred
constexpr size_t OFF_HALF = OFF_FLAT + sizeof(_Float16) * (size_t)B * N * H;

DEV int laneid() { return (int)(threadIdx.x & 31u); }

// A-matrix tile (16x32, f16): lane L<16 row M=L holds K {kb..kb+7, kb+16..kb+23},
// lanes 16..31 hold K shifted by +8 (ISA 7.12.2 16-bit A layout).
DEV v16h a_tile_f16(const _Float16* base, int stride, int mbase, int kbase) {
  int l  = laneid();
  int m  = mbase + (l & 15);
  int k0 = kbase + ((l >> 4) << 3);
  const _Float16* p = base + (size_t)m * stride + k0;
  v8h lo = *(const v8h*)(p);
  v8h hi = *(const v8h*)(p + 16);
  return __builtin_shufflevector(lo, hi, 0,1,2,3,4,5,6,7,8,9,10,11,12,13,14,15);
}

// Same A layout, sourced from f32 global memory with cvt + row guard.
DEV v16h a_tile_f32(const float* base, int stride, int mbase, int kbase, int mlimit) {
  int l  = laneid();
  int m  = mbase + (l & 15);
  int k0 = kbase + ((l >> 4) << 3);
  const float* p = base + (size_t)m * stride;
  bool ok = (m < mlimit);
  v16h a;
#pragma unroll
  for (int t = 0; t < 8; ++t) a[t]     = (_Float16)(ok ? p[k0 + t]      : 0.0f);
#pragma unroll
  for (int t = 0; t < 8; ++t) a[t + 8] = (_Float16)(ok ? p[k0 + 16 + t] : 0.0f);
  return a;
}

// B-matrix tile (32x16, f16) from transposed weights WT[n][k]:
// lane L<16 holds column n=L, K {kb..kb+7, kb+16..kb+23}; lanes 16..31 hold K shifted by +8.
DEV v16h b_tile_f16(const _Float16* baseT, int stride, int nbase, int kbase) {
  int l  = laneid();
  int n  = nbase + (l & 15);
  int k0 = kbase + ((l >> 4) << 3);
  const _Float16* p = baseT + (size_t)n * stride + k0;
  v8h lo = *(const v8h*)(p);
  v8h hi = *(const v8h*)(p + 16);
  return __builtin_shufflevector(lo, hi, 0,1,2,3,4,5,6,7,8,9,10,11,12,13,14,15);
}

DEV v8f wmma(v16h a, v16h b, v8f c) {
  return __builtin_amdgcn_wmma_f32_16x16x32_f16(false, a, false, b, (short)0, c, false, false);
}

DEV float softplusf_(float x) { return fmaxf(x, 0.0f) + log1pf(expf(-fabsf(x))); }

// CDNA5 async copy: global -> LDS, 16B per lane, tracked by ASYNCcnt.
DEV void async_g2l_b128(uint32_t lds_byte_off, const void* gaddr) {
  asm volatile("global_load_async_to_lds_b128 %0, %1, off"
               :: "v"(lds_byte_off), "v"(gaddr) : "memory");
}
DEV void wait_asynccnt0() { asm volatile("s_wait_asynccnt 0x0" ::: "memory"); }

// ---------------- weight conversion / transposition to f16 ----------------
__global__ void k_convert(const float* fc1w, const float* fc2w,
                          const float* whr, const float* whi, const float* whh,
                          const float* wir, const float* wii, const float* win,
                          const float* wm1, const float* wm2,
                          const float* wdf, const float* wloc, const float* wscale,
                          _Float16* hb) {
  int idx = blockIdx.x * blockDim.x + threadIdx.x;
  if (idx < 8192) {  // W1catT[c][k]: c<64 -> W1[k][c] (RECV), c>=64 -> W1[64+k][c-64] (SEND); edge type 1
    int c = idx >> 6, k = idx & 63;
    int kk = (c < 64) ? k : (64 + k);
    hb[HO_W1 + idx] = (_Float16)fc1w[(size_t)(2 * H * H) + (size_t)kk * H + (c & 63)];
    return;
  }
  idx -= 8192;
  if (idx < 4096) { int n = idx >> 6, k = idx & 63; hb[HO_W2  + idx] = (_Float16)fc2w[(size_t)(H * H) + (size_t)k * H + n]; return; }
  idx -= 4096;
  if (idx < 4096) { int n = idx >> 6, k = idx & 63; hb[HO_WHR + idx] = (_Float16)whr[(size_t)k * H + n]; return; }
  idx -= 4096;
  if (idx < 4096) { int n = idx >> 6, k = idx & 63; hb[HO_WHI + idx] = (_Float16)whi[(size_t)k * H + n]; return; }
  idx -= 4096;
  if (idx < 4096) { int n = idx >> 6, k = idx & 63; hb[HO_WHH + idx] = (_Float16)whh[(size_t)k * H + n]; return; }
  idx -= 4096;
  if (idx < 2048) { int n = idx >> 5, k = idx & 31; hb[HO_WIR + idx] = (_Float16)wir[(size_t)k * H + n]; return; }
  idx -= 2048;
  if (idx < 2048) { int n = idx >> 5, k = idx & 31; hb[HO_WII + idx] = (_Float16)wii[(size_t)k * H + n]; return; }
  idx -= 2048;
  if (idx < 2048) { int n = idx >> 5, k = idx & 31; hb[HO_WIN + idx] = (_Float16)win[(size_t)k * H + n]; return; }
  idx -= 2048;
  if (idx < 4096) { int n = idx >> 6, k = idx & 63; hb[HO_WM1 + idx] = (_Float16)wm1[(size_t)k * H + n]; return; }
  idx -= 4096;
  if (idx < 4096) { int n = idx >> 6, k = idx & 63; hb[HO_WM2 + idx] = (_Float16)wm2[(size_t)k * H + n]; return; }
  idx -= 4096;
  if (idx < 3 * NPAD * (N * H)) {   // projT[o][n][k], pad n in [100,112) with zeros
    int o = idx / (NPAD * N * H);
    int rem = idx % (NPAD * N * H);
    int n = rem / (N * H);
    int k = rem % (N * H);
    const float* w = (o == 0) ? wdf : (o == 1) ? wloc : wscale;
    hb[HO_PROJ + idx] = (_Float16)((n < N) ? w[(size_t)k * N + n] : 0.0f);
  }
}

// ---------------- hard gumbel-softmax (forward) + type-1 gate ----------------
__global__ void k_edges(const float* logits, const float* noise, float* edges_out, float* gate) {
  int idx = blockIdx.x * blockDim.x + threadIdx.x;
  if (idx >= B * E) return;
  float z0 = (logits[2 * idx]     + noise[2 * idx])     * 2.0f;   // 1/TAU = 2
  float z1 = (logits[2 * idx + 1] + noise[2 * idx + 1]) * 2.0f;
  float mx = fmaxf(z0, z1);
  float e0 = expf(z0 - mx), e1 = expf(z1 - mx);
  float inv = 1.0f / (e0 + e1);
  float y0 = e0 * inv, y1 = e1 * inv;
  float h0 = (z0 >= z1) ? 1.0f : 0.0f;   // argmax, first-wins like jnp.argmax
  float h1 = 1.0f - h0;
  float o0 = (h0 - y0) + y0;             // straight-through forward value
  float o1 = (h1 - y1) + y1;
  edges_out[2 * idx] = o0;
  edges_out[2 * idx + 1] = o1;
  gate[idx] = o1;
}

// ---------------- per-batch layer-1 split:  X = H @ W1_bot, Y = H @ W1_top + b1 ----------------
__global__ void __launch_bounds__(128) k_xy(const float* hidden, const float* fc1b,
                                            const _Float16* w1catT, float* X, float* Y) {
  int b = blockIdx.x;
  int wave = (int)(threadIdx.x >> 5);
  const float* hb = hidden + (size_t)b * (N * H);
  // Hoist all 16 B-tiles of W1catT: avoids a loadcnt-0 stall in front of every wmma.
  v16h Bt[8][2];
#pragma unroll
  for (int nt = 0; nt < 8; ++nt) {
    Bt[nt][0] = b_tile_f16(w1catT, 64, nt * 16, 0);
    Bt[nt][1] = b_tile_f16(w1catT, 64, nt * 16, 32);
  }
  for (int mt = wave; mt < 7; mt += 4) {
    v16h A0 = a_tile_f32(hb, H, mt * 16, 0,  N);
    v16h A1 = a_tile_f32(hb, H, mt * 16, 32, N);
#pragma unroll
    for (int nt = 0; nt < 8; ++nt) {
      v8f c = {};
      c = wmma(A0, Bt[nt][0], c);
      c = wmma(A1, Bt[nt][1], c);
      int l = laneid();
      int ncol = nt * 16 + (l & 15);
      int hi = l >> 4;
      float b1v = (ncol < 64) ? fc1b[64 + ncol] : 0.0f;  // msg_fc1_b[1]
#pragma unroll
      for (int r = 0; r < 8; ++r) {
        int m = mt * 16 + r + 8 * hi;
        if (ncol < 64) {
          if (m < N) Y[(size_t)b * (N * H) + (size_t)m * H + ncol] = c[r] + b1v;
        } else {
          X[(size_t)b * (NPAD * H) + (size_t)m * H + (ncol - 64)] = (m < N) ? c[r] : 0.0f;
        }
      }
    }
  }
}

// ---------------- heavy kernel: per (b, receiver j) messages + gated reduction ----------------
__global__ void __launch_bounds__(128) k_msg(const float* Xall, const float* Yall,
                                             const float* gate, const _Float16* w2T,
                                             const float* fc2b, float* agg) {
  __shared__ float    xs[NPAD * H];        // raw X tile, filled by async DMA (28 KB)
  __shared__ _Float16 m1[NPAD * LSTR];     // tanh(X_i + Y_j) in f16, padded stride
  __shared__ float yj[H];
  __shared__ float gj[NPAD];
  __shared__ float aggl[H];
  int b = blockIdx.x / N;
  int j = blockIdx.x % N;
  int tid = (int)threadIdx.x;

  // Kick off async global->LDS copy of X[b] (1792 x 16B chunks, 14 per thread).
  {
    const float* Xb = Xall + (size_t)b * (NPAD * H);
    uint32_t lbase = (uint32_t)(uintptr_t)(void*)xs;
#pragma unroll
    for (int c = 0; c < (NPAD * H) / (4 * 128); ++c) {
      int chunk = tid + c * 128;
      async_g2l_b128(lbase + (uint32_t)(chunk * 16), (const void*)(Xb + chunk * 4));
    }
  }
  // Stage Y_j, gates, zero the reduction buffer while the DMA is in flight.
  if (tid < H) { yj[tid] = Yall[(size_t)b * (N * H) + (size_t)j * H + tid]; aggl[tid] = 0.0f; }
  if (tid < NPAD) {
    float g = 0.0f;
    if (tid < N && tid != j) {                 // sender i = tid, edge (i -> j)
      int e = tid * 99 + j - ((j > tid) ? 1 : 0);
      g = gate[(size_t)b * E + e];
    }
    gj[tid] = g;
  }
  wait_asynccnt0();
  __syncthreads();

  for (int idx = tid; idx < NPAD * H; idx += 128) {   // m1 = tanh(X_i + Y_j), f16 into LDS
    int k = idx & 63;
    m1[(idx >> 6) * LSTR + k] = (_Float16)tanhf(xs[idx] + yj[k]);
  }
  __syncthreads();

  int wave = tid >> 5;
  int l = laneid();
  int hi = l >> 4;
  v16h Bt[4][2];
#pragma unroll
  for (int nt = 0; nt < 4; ++nt) {
    Bt[nt][0] = b_tile_f16(w2T, 64, nt * 16, 0);
    Bt[nt][1] = b_tile_f16(w2T, 64, nt * 16, 32);
  }
  float b2v[4];
#pragma unroll
  for (int nt = 0; nt < 4; ++nt) b2v[nt] = fc2b[64 + nt * 16 + (l & 15)];  // msg_fc2_b[1]
  float acc[4] = {0.f, 0.f, 0.f, 0.f};

  for (int mt = wave; mt < 7; mt += 4) {
    v16h A0 = a_tile_f16(m1, LSTR, mt * 16, 0);
    v16h A1 = a_tile_f16(m1, LSTR, mt * 16, 32);
#pragma unroll
    for (int nt = 0; nt < 4; ++nt) {
      v8f c = {};
      c = wmma(A0, Bt[nt][0], c);
      c = wmma(A1, Bt[nt][1], c);
#pragma unroll
      for (int r = 0; r < 8; ++r) {
        int m = mt * 16 + r + 8 * hi;                 // sender index
        if (m < N) acc[nt] += tanhf(c[r] + b2v[nt]) * gj[m];
      }
    }
  }
#pragma unroll
  for (int nt = 0; nt < 4; ++nt) atomicAdd(&aggl[nt * 16 + (l & 15)], acc[nt]);
  __syncthreads();
  if (tid < H) agg[(size_t)b * (N * H) + (size_t)j * H + tid] = aggl[tid] * (1.0f / 99.0f);
}

// ---------------- GRU update + 2-layer output MLP (16-row tiles) ----------------
__global__ void __launch_bounds__(32) k_gru(const float* agg, const float* inputs, const float* hidden,
    const _Float16* whrT, const _Float16* whiT, const _Float16* whhT,
    const _Float16* wirT, const _Float16* wiiT, const _Float16* winT,
    const _Float16* wm1T, const _Float16* wm2T,
    const float* bir, const float* bii, const float* bin,
    const float* mb1, const float* mb2,
    float* nh_out, _Float16* flat_h) {
  __shared__ _Float16 hl[16 * LSTR];
  __shared__ _Float16 pl[16 * LSTR];
  int R0 = blockIdx.x * 16;
  int l = laneid();
  int hi = l >> 4;
  int nloc = l & 15;
  v16h Aagg0 = a_tile_f32(agg, H, R0, 0,  B * N);
  v16h Aagg1 = a_tile_f32(agg, H, R0, 32, B * N);
  v16h Ain   = a_tile_f32(inputs, IN, R0, 0, B * N);
  for (int nt = 0; nt < 4; ++nt) {
    v8f cr = {}, ci = {}, cnx = {}, cnh = {};
    cr  = wmma(Aagg0, b_tile_f16(whrT, 64, nt * 16, 0),  cr);
    cr  = wmma(Aagg1, b_tile_f16(whrT, 64, nt * 16, 32), cr);
    ci  = wmma(Aagg0, b_tile_f16(whiT, 64, nt * 16, 0),  ci);
    ci  = wmma(Aagg1, b_tile_f16(whiT, 64, nt * 16, 32), ci);
    cnh = wmma(Aagg0, b_tile_f16(whhT, 64, nt * 16, 0),  cnh);
    cnh = wmma(Aagg1, b_tile_f16(whhT, 64, nt * 16, 32), cnh);
    cr  = wmma(Ain, b_tile_f16(wirT, 32, nt * 16, 0), cr);
    ci  = wmma(Ain, b_tile_f16(wiiT, 32, nt * 16, 0), ci);
    cnx = wmma(Ain, b_tile_f16(winT, 32, nt * 16, 0), cnx);
    int n = nt * 16 + nloc;
    float birv = bir[n], biiv = bii[n], binv = bin[n];
#pragma unroll
    for (int r = 0; r < 8; ++r) {
      int m = R0 + r + 8 * hi;
      float hold = hidden[(size_t)m * H + n];
      float rg = 1.0f / (1.0f + expf(-(cr[r] + birv)));
      float ig = 1.0f / (1.0f + expf(-(ci[r] + biiv)));
      float ng = tanhf(cnx[r] + binv + rg * cnh[r]);
      float hn = (1.0f - ig) * ng + ig * hold;
      nh_out[(size_t)m * H + n] = hn;
      hl[(r + 8 * hi) * LSTR + n] = (_Float16)hn;
    }
  }
  __syncthreads();
  for (int nt = 0; nt < 4; ++nt) {
    v8f c = {};
    c = wmma(a_tile_f16(hl, LSTR, 0, 0),  b_tile_f16(wm1T, 64, nt * 16, 0),  c);
    c = wmma(a_tile_f16(hl, LSTR, 0, 32), b_tile_f16(wm1T, 64, nt * 16, 32), c);
    int n = nt * 16 + nloc;
    float bv = mb1[n];
#pragma unroll
    for (int r = 0; r < 8; ++r)
      pl[(r + 8 * hi) * LSTR + n] = (_Float16)fmaxf(c[r] + bv, 0.0f);
  }
  __syncthreads();
  for (int nt = 0; nt < 4; ++nt) {
    v8f c = {};
    c = wmma(a_tile_f16(pl, LSTR, 0, 0),  b_tile_f16(wm2T, 64, nt * 16, 0),  c);
    c = wmma(a_tile_f16(pl, LSTR, 0, 32), b_tile_f16(wm2T, 64, nt * 16, 32), c);
    int n = nt * 16 + nloc;
    float bv = mb2[n];
#pragma unroll
    for (int r = 0; r < 8; ++r) {
      int m = R0 + r + 8 * hi;
      float v = fmaxf(c[r] + bv, 0.0f);
      flat_h[(size_t)(m / N) * (N * H) + (size_t)(m % N) * H + n] = (_Float16)v;
    }
  }
}

// ---------------- StudentT arg projections: flat[64,6400] @ W[6400,100] x3 ----------------
__global__ void __launch_bounds__(32) k_proj(const _Float16* flat_h, const _Float16* projT,
                                             const float* bdf, const float* bloc, const float* bscale,
                                             float* out) {
  int bx = blockIdx.x;
  int o  = bx / 28;
  int mt = (bx % 28) / 7;
  int nt = bx % 7;
  const _Float16* W = projT + (size_t)o * NPAD * (N * H);
  int l = laneid();
  v8f c = {};
  for (int kb = 0; kb < N * H; kb += 32) {
    if (kb + 512 < N * H) {   // stream ahead (global_prefetch_b8)
      __builtin_prefetch(flat_h + (size_t)(mt * 16 + (l & 15)) * (N * H) + kb + 512, 0, 1);
      __builtin_prefetch(W + (size_t)(nt * 16 + (l & 15)) * (N * H) + kb + 512, 0, 1);
    }
    v16h a  = a_tile_f16(flat_h, N * H, mt * 16, kb);
    v16h bm = b_tile_f16(W, N * H, nt * 16, kb);
    c = wmma(a, bm, c);
  }
  int n = nt * 16 + (l & 15);
  if (n >= N) return;
  const float* bias = (o == 0) ? bdf : (o == 1) ? bloc : bscale;
  float bv = bias[n];
  int hi = l >> 4;
#pragma unroll
  for (int r = 0; r < 8; ++r) {
    int m = mt * 16 + r + 8 * hi;   // batch row (< 64)
    float x = c[r] + bv;
    float v = (o == 0) ? (2.0f + softplusf_(x)) : (o == 1) ? x : softplusf_(x);
    out[(size_t)o * (B * N) + (size_t)m * N + n] = v;
  }
}

extern "C" void kernel_launch(void* const* d_in, const int* in_sizes, int n_in,
                              void* d_out, int out_size, void* d_ws, size_t ws_size,
                              hipStream_t stream) {
  const float* inputs = (const float*)d_in[0];
  const float* hidden = (const float*)d_in[1];
  const float* logits = (const float*)d_in[2];
  const float* noise  = (const float*)d_in[3];
  const float* fc1w   = (const float*)d_in[4];
  const float* fc1b   = (const float*)d_in[5];
  const float* fc2w   = (const float*)d_in[6];
  const float* fc2b   = (const float*)d_in[7];
  const float* whr    = (const float*)d_in[8];
  const float* whi    = (const float*)d_in[9];
  const float* whh    = (const float*)d_in[10];
  const float* wir    = (const float*)d_in[11];
  const float* bir    = (const float*)d_in[12];
  const float* wii    = (const float*)d_in[13];
  const float* bii    = (const float*)d_in[14];
  const float* win    = (const float*)d_in[15];
  const float* bin    = (const float*)d_in[16];
  const float* wm1    = (const float*)d_in[17];
  const float* mb1    = (const float*)d_in[18];
  const float* wm2    = (const float*)d_in[19];
  const float* mb2    = (const float*)d_in[20];
  const float* wdf    = (const float*)d_in[21];
  const float* bdf    = (const float*)d_in[22];
  const float* wloc   = (const float*)d_in[23];
  const float* bloc   = (const float*)d_in[24];
  const float* wscale = (const float*)d_in[25];
  const float* bscale = (const float*)d_in[26];

  float* out       = (float*)d_out;
  float* out_nh    = out + 3 * B * N;                 // new_hidden after df/loc/scale
  float* out_edges = out + 3 * B * N + B * N * H;     // edges last

  char* ws = (char*)d_ws;
  float*    gate   = (float*)(ws + OFF_GATE);
  float*    X      = (float*)(ws + OFF_X);
  float*    Y      = (float*)(ws + OFF_Y);
  float*    agg    = (float*)(ws + OFF_AGG);
  _Float16* flat_h = (_Float16*)(ws + OFF_FLAT);
  _Float16* hbase  = (_Float16*)(ws + OFF_HALF);

  k_convert<<<(H_TOTAL + 255) / 256, 256, 0, stream>>>(fc1w, fc2w, whr, whi, whh,
                                                       wir, wii, win, wm1, wm2,
                                                       wdf, wloc, wscale, hbase);
  k_edges<<<(B * E + 255) / 256, 256, 0, stream>>>(logits, noise, out_edges, gate);
  k_xy<<<B, 128, 0, stream>>>(hidden, fc1b, hbase + HO_W1, X, Y);
  k_msg<<<B * N, 128, 0, stream>>>(X, Y, gate, hbase + HO_W2, fc2b, agg);
  k_gru<<<(B * N) / 16, 32, 0, stream>>>(agg, inputs, hidden,
                                         hbase + HO_WHR, hbase + HO_WHI, hbase + HO_WHH,
                                         hbase + HO_WIR, hbase + HO_WII, hbase + HO_WIN,
                                         hbase + HO_WM1, hbase + HO_WM2,
                                         bir, bii, bin, mb1, mb2, out_nh, flat_h);
  k_proj<<<84, 32, 0, stream>>>(flat_h, hbase + HO_PROJ, bdf, bloc, bscale, out);
}
